// TraFixV5_42365557408075
// MI455X (gfx1250) — matrix operand order repaired and checked by reading, not compiled, not run on hardware
//
#include <hip/hip_runtime.h>
#include <hip/hip_bf16.h>

// ---------------------------------------------------------------------------
// Problem constants (from reference): B=1024, T=128, J=5, D=64, H=128
//   BJ = 5120 GRU rows, 3H = 384 gate columns, GAT_OUT = MID = 128, OUT = 64
// ---------------------------------------------------------------------------

typedef __attribute__((ext_vector_type(16))) __bf16 v16bf;
typedef __attribute__((ext_vector_type(2)))  __bf16 bf16x2;
typedef __attribute__((ext_vector_type(8)))  float  v8f;

// 16-bit A-matrix 16x32 K map (CDNA5 ISA §7.12.2):
// lanes 0-15: v0:K0,1 v1:K2,3 v2:K4,5 v3:K6,7 v4:K16,17.. ; lanes16-31: +8/+24
__device__ __forceinline__ int kmapA(int lane, int e) {
  int v = e >> 1;
  return ((lane & 16) ? 8 : 0) + ((v & 3) << 1) + ((v >> 2) << 4) + (e & 1);
}
// 16-bit B-matrix 32x16: same K map as A (lanes 0-15 K0..7,16..23; lanes 16-31 K8..15,24..31)
__device__ __forceinline__ int kmapB(int lane, int e) {
  return kmapA(lane, e);
}

// ---------------------------------------------------------------------------
// Kernel 0: convert weights f32 -> bf16 into workspace (once per launch)
// ---------------------------------------------------------------------------
__global__ void __launch_bounds__(256)
prep_kernel(const float* __restrict__ Wih, const float* __restrict__ Whh,
            const float* __restrict__ gW,  const float* __restrict__ W1,
            const float* __restrict__ W2,
            __bf16* __restrict__ oWih, __bf16* __restrict__ oWhh,
            __bf16* __restrict__ ogW,  __bf16* __restrict__ oW1,
            __bf16* __restrict__ oW2)
{
  int tid = blockIdx.x * 256 + threadIdx.x;
  if (tid < 24576)            oWih[tid]          = (__bf16)Wih[tid];
  else if (tid < 73728)       oWhh[tid - 24576]  = (__bf16)Whh[tid - 24576];
  else if (tid < 90112)       ogW [tid - 73728]  = (__bf16)gW [tid - 73728];
  else if (tid < 106496)      oW1 [tid - 90112]  = (__bf16)W1 [tid - 90112];
  else if (tid < 114688)      oW2 [tid - 106496] = (__bf16)W2 [tid - 106496];
}

// ---------------------------------------------------------------------------
// Kernel 1: GRU over T=128 steps. One block = 16 rows of BJ=5120, 8 waves.
// Wave w owns gate-column tiles {3w, 3w+1, 3w+2} of 24 (384 cols / 16).
// Weight B-fragments stationary in VGPRs; h kept in LDS both as f32 state
// and as packed bf16 pairs (direct WMMA A-fragment feed, no per-step cvt).
// ---------------------------------------------------------------------------
#define LDG 388   // padded pitch for 384-wide gate tiles (bank-conflict free)
#define LDH 132   // padded pitch for 128-wide f32 h tile
#define LDP 66    // padded pitch for 64-wide packed-bf16 h tile

__global__ void __launch_bounds__(256)
gru_kernel(const float* __restrict__ obs,
           const __bf16* __restrict__ Wih,   // [384*64]  bf16
           const __bf16* __restrict__ Whh,   // [384*128] bf16
           const float* __restrict__ bih,    // [384]
           const float* __restrict__ bhh,    // [384]
           float*  __restrict__ hOut,        // [5120*128]
           __bf16* __restrict__ hOutBf)      // [5120*128]
{
  __shared__ float        sGI[16][LDG];
  __shared__ float        sGH[16][LDG];
  __shared__ float        sH [16][LDH];
  __shared__ unsigned int sHb[16][LDP];   // packed bf16 pairs of h
  __shared__ float        sBI[384];
  __shared__ float        sBH[384];

  const int tid   = threadIdx.x;
  const int lane  = tid & 31;
  const int wave  = tid >> 5;
  const int nLane = lane & 15;
  const int rowBase = blockIdx.x * 16;

  for (int i = tid; i < 16 * 128; i += 256) sH[i >> 7][i & 127] = 0.f;
  for (int i = tid; i < 16 * 64; i += 256) sHb[i >> 6][i & 63] = 0u;
  for (int i = tid; i < 384; i += 256) { sBI[i] = bih[i]; sBH[i] = bhh[i]; }

  // Stationary B fragments: W_ih^T (K=64 -> 2 frags) and W_hh^T (K=128 -> 4)
  v16bf bI[3][2];
  v16bf bH[3][4];
#pragma unroll
  for (int s = 0; s < 3; ++s) {
    const int c = (3 * wave + s) * 16 + nLane;          // gate column 0..383
#pragma unroll
    for (int kf = 0; kf < 2; ++kf)
#pragma unroll
      for (int e = 0; e < 16; ++e)
        bI[s][kf][e] = Wih[c * 64 + kf * 32 + kmapB(lane, e)];
#pragma unroll
    for (int kf = 0; kf < 4; ++kf)
#pragma unroll
      for (int e = 0; e < 16; ++e)
        bH[s][kf][e] = Whh[c * 128 + kf * 32 + kmapB(lane, e)];
  }

  // x row for this lane: n = rowBase + nLane; x[n,t,d] = obs[((b*T+t)*J+j)*D+d]
  const int myRow = rowBase + nLane;
  const int bIdx = myRow / 5, jIdx = myRow % 5;
  const float* xRow = obs + ((size_t)bIdx * 128 * 5 + jIdx) * 64;

  __syncthreads();

#pragma unroll 1
  for (int t = 0; t < 128; ++t) {
    const float* xr = xRow + (size_t)t * 5 * 64;
    __builtin_prefetch(xr + 5 * 64, 0, 1);   // next timestep -> global_prefetch

    // A fragments: x from global f32 (convert), h from LDS packed bf16 (no cvt)
    v16bf aX[2], aH[4];
#pragma unroll
    for (int kf = 0; kf < 2; ++kf)
#pragma unroll
      for (int e = 0; e < 16; ++e)
        aX[kf][e] = (__bf16)xr[kf * 32 + kmapA(lane, e)];
#pragma unroll
    for (int kf = 0; kf < 4; ++kf)
#pragma unroll
      for (int p = 0; p < 8; ++p) {
        const unsigned int u = sHb[nLane][(kf * 32 + kmapA(lane, 2 * p)) >> 1];
        const bf16x2 hp = __builtin_bit_cast(bf16x2, u);
        aH[kf][2 * p]     = hp.x;
        aH[kf][2 * p + 1] = hp.y;
      }

#pragma unroll
    for (int s = 0; s < 3; ++s) {
      v8f accI = {};
      v8f accH = {};
#pragma unroll
      for (int kf = 0; kf < 2; ++kf)
        accI = __builtin_amdgcn_wmma_f32_16x16x32_bf16(
            false, aX[kf], false, bI[s][kf], (short)0, accI, false, false);
#pragma unroll
      for (int kf = 0; kf < 4; ++kf)
        accH = __builtin_amdgcn_wmma_f32_16x16x32_bf16(
            false, aH[kf], false, bH[s][kf], (short)0, accH, false, false);

      const int colT = (3 * wave + s) * 16 + nLane;
      const int rOff = (lane >> 4) * 8;        // C/D layout: M = r + 8*(lane/16)
#pragma unroll
      for (int r = 0; r < 8; ++r) {
        sGI[rOff + r][colT] = accI[r];
        sGH[rOff + r][colT] = accH[r];
      }
    }
    __syncthreads();

    // Fused gates: r = sig(i_r+h_r), z = sig(i_z+h_z), n = tanh(i_n + r*h_n)
    {
      const int row = tid & 15;
      const int c0  = (tid >> 4) * 8;
      float hv[8];
#pragma unroll
      for (int q = 0; q < 8; ++q) {
        const int c = c0 + q;
        float iR = sGI[row][c]       + sBI[c];
        float hR = sGH[row][c]       + sBH[c];
        float iZ = sGI[row][c + 128] + sBI[c + 128];
        float hZ = sGH[row][c + 128] + sBH[c + 128];
        float iN = sGI[row][c + 256] + sBI[c + 256];
        float hN = sGH[row][c + 256] + sBH[c + 256];
        float rg = 1.f / (1.f + __expf(-(iR + hR)));
        float zg = 1.f / (1.f + __expf(-(iZ + hZ)));
        float ng = tanhf(iN + rg * hN);
        hv[q] = (1.f - zg) * ng + zg * sH[row][c];
        sH[row][c] = hv[q];
      }
#pragma unroll
      for (int q = 0; q < 8; q += 2) {
        bf16x2 hp;
        hp.x = (__bf16)hv[q];
        hp.y = (__bf16)hv[q + 1];
        sHb[row][(c0 + q) >> 1] = __builtin_bit_cast(unsigned int, hp);
      }
    }
    __syncthreads();
  }

  for (int i = tid; i < 16 * 128; i += 256) {
    const int r = i >> 7, c = i & 127;
    const float v = sH[r][c];
    const size_t o = (size_t)(rowBase + r) * 128 + c;
    hOut[o] = v;
    hOutBf[o] = (__bf16)v;
  }
}

// ---------------------------------------------------------------------------
// Kernel 2: generic bf16 WMMA GEMM  C[M,N] = A[M,K] * B[K,N] (+bias)(+relu)
// One wave = one 16x16 C tile; block = 4 waves.
// ---------------------------------------------------------------------------
__global__ void __launch_bounds__(128)
gemm_bf16_kernel(const __bf16* __restrict__ A, const __bf16* __restrict__ Bm,
                 const float* __restrict__ bias, float* __restrict__ C,
                 __bf16* __restrict__ Cbf, int M, int N, int K, int relu)
{
  const int lane = threadIdx.x & 31;
  const int wave = threadIdx.x >> 5;
  const int tile = blockIdx.x * 4 + wave;
  const int tilesN = N >> 4;
  const int tm = tile / tilesN, tn = tile % tilesN;
  if (tm * 16 >= M) return;                 // wave-uniform; EXEC stays full

  const int row = tm * 16 + (lane & 15);
  const int col = tn * 16 + (lane & 15);
  v8f acc = {};
  for (int kf = 0; kf < (K >> 5); ++kf) {
    v16bf a, b;
#pragma unroll
    for (int e = 0; e < 16; ++e) {
      a[e] = A[(size_t)row * K + kf * 32 + kmapA(lane, e)];
      b[e] = Bm[(size_t)(kf * 32 + kmapB(lane, e)) * N + col];
    }
    acc = __builtin_amdgcn_wmma_f32_16x16x32_bf16(
        false, a, false, b, (short)0, acc, false, false);
  }
  const int rOff = (lane >> 4) * 8;
  const float bv = bias ? bias[col] : 0.f;
#pragma unroll
  for (int r = 0; r < 8; ++r) {
    float v = acc[r] + bv;
    if (relu) v = fmaxf(v, 0.f);
    const size_t o = (size_t)(tm * 16 + rOff + r) * N + col;
    C[o] = v;
    if (Cbf) Cbf[o] = (__bf16)v;
  }
}

// ---------------------------------------------------------------------------
// Kernel 3: GAT attention over the fixed 5-node chain (+self loops).
// Thread per (batch, dst node i, head). adj(i,j) <=> |i-j| <= 1.
// ---------------------------------------------------------------------------
__global__ void __launch_bounds__(256)
gat_attn_kernel(const float* __restrict__ xp, const float* __restrict__ attS,
                const float* __restrict__ attD, const float* __restrict__ gatB,
                float* __restrict__ g, __bf16* __restrict__ gbf)
{
  const int tid = blockIdx.x * 256 + threadIdx.x;
  if (tid >= 1024 * 5 * 4) return;
  const int head = tid & 3;
  const int i    = (tid >> 2) % 5;
  const int b    = tid / 20;
  const float* aS = attS + head * 32;
  const float* aD = attD + head * 32;

  const float* xi = xp + (size_t)(b * 5 + i) * 128 + head * 32;
  float ad = 0.f;
  for (int d = 0; d < 32; ++d) ad += xi[d] * aD[d];

  float sc[5];
  float mx = -1e30f;
  for (int j = 0; j < 5; ++j) {
    const int dj = j - i;
    if (dj < -1 || dj > 1) { sc[j] = -1e30f; continue; }
    const float* xj = xp + (size_t)(b * 5 + j) * 128 + head * 32;
    float as = 0.f;
    for (int d = 0; d < 32; ++d) as += xj[d] * aS[d];
    float e = as + ad;
    e = (e > 0.f) ? e : 0.2f * e;            // leaky_relu(0.2)
    sc[j] = e;
    mx = fmaxf(mx, e);
  }
  float den = 0.f;
  for (int j = 0; j < 5; ++j) {
    sc[j] = (sc[j] > -1e29f) ? __expf(sc[j] - mx) : 0.f;
    den += sc[j];
  }
  const float inv = 1.f / den;

  float outv[32];
  for (int d = 0; d < 32; ++d) outv[d] = 0.f;
  for (int j = 0; j < 5; ++j) {
    const float al = sc[j] * inv;
    if (al == 0.f) continue;
    const float* xj = xp + (size_t)(b * 5 + j) * 128 + head * 32;
    for (int d = 0; d < 32; ++d) outv[d] += al * xj[d];
  }
  const size_t o = (size_t)(b * 5 + i) * 128 + head * 32;
  for (int d = 0; d < 32; ++d) {
    const float v = outv[d] + gatB[head * 32 + d];
    g[o + d] = v;
    gbf[o + d] = (__bf16)v;
  }
}

// ---------------------------------------------------------------------------
// Kernel 4: per-junction actor heads + critic on junction-mean.
// d_out = [logits 1024*5*4][value 1024]
// ---------------------------------------------------------------------------
__global__ void __launch_bounds__(256)
heads_kernel(const float* __restrict__ tf, const float* __restrict__ aW,
             const float* __restrict__ aB, const float* __restrict__ cW,
             const float* __restrict__ cB, float* __restrict__ out)
{
  const int tid = blockIdx.x * 256 + threadIdx.x;
  if (tid < 1024 * 5 * 4) {
    const int p = tid & 3;
    const int j = (tid >> 2) % 5;
    const int b = tid / 20;
    const float* tr = tf + (size_t)(b * 5 + j) * 64;
    float acc = aB[j * 4 + p];
    for (int o = 0; o < 64; ++o) acc += tr[o] * aW[(j * 64 + o) * 4 + p];
    out[(size_t)(b * 5 + j) * 4 + p] = acc;
  } else if (tid < 1024 * 5 * 4 + 1024) {
    const int b = tid - 1024 * 5 * 4;
    float acc = 0.f;
    for (int j = 0; j < 5; ++j) {
      const float* tr = tf + (size_t)(b * 5 + j) * 64;
      for (int o = 0; o < 64; ++o) acc += tr[o] * cW[o];
    }
    out[20480 + b] = acc * 0.2f + cB[0];
  }
}

// ---------------------------------------------------------------------------
extern "C" void kernel_launch(void* const* d_in, const int* in_sizes, int n_in,
                              void* d_out, int out_size, void* d_ws, size_t ws_size,
                              hipStream_t stream) {
  const float* obs  = (const float*)d_in[0];
  // d_in[1] = edge_index: fixed bidirectional 5-chain, adjacency hardcoded
  const float* Wih  = (const float*)d_in[2];
  const float* Whh  = (const float*)d_in[3];
  const float* bih  = (const float*)d_in[4];
  const float* bhh  = (const float*)d_in[5];
  const float* gW   = (const float*)d_in[6];
  const float* attS = (const float*)d_in[7];
  const float* attD = (const float*)d_in[8];
  const float* gB   = (const float*)d_in[9];
  const float* W1   = (const float*)d_in[10];
  const float* b1   = (const float*)d_in[11];
  const float* W2   = (const float*)d_in[12];
  const float* b2   = (const float*)d_in[13];
  const float* aW   = (const float*)d_in[14];
  const float* aB   = (const float*)d_in[15];
  const float* cW   = (const float*)d_in[16];
  const float* cB   = (const float*)d_in[17];

  char* w = (char*)d_ws;
  size_t off = 0;
  __bf16* oWih = (__bf16*)(w + off); off += 24576 * 2;
  __bf16* oWhh = (__bf16*)(w + off); off += 49152 * 2;
  __bf16* ogW  = (__bf16*)(w + off); off += 16384 * 2;
  __bf16* oW1  = (__bf16*)(w + off); off += 16384 * 2;
  __bf16* oW2  = (__bf16*)(w + off); off += 8192 * 2;
  float*  hF   = (float*) (w + off); off += (size_t)5120 * 128 * 4;
  __bf16* hBf  = (__bf16*)(w + off); off += (size_t)5120 * 128 * 2;
  float*  xpF  = (float*) (w + off); off += (size_t)5120 * 128 * 4;
  float*  gF   = (float*) (w + off); off += (size_t)5120 * 128 * 4;
  __bf16* gBf  = (__bf16*)(w + off); off += (size_t)5120 * 128 * 2;
  float*  mF   = (float*) (w + off); off += (size_t)5120 * 128 * 4;
  __bf16* mBf  = (__bf16*)(w + off); off += (size_t)5120 * 128 * 2;
  float*  tF   = (float*) (w + off); off += (size_t)5120 * 64 * 4;

  // 0) weights -> bf16
  prep_kernel<<<448, 256, 0, stream>>>(Wih, Whh, gW, W1, W2,
                                       oWih, oWhh, ogW, oW1, oW2);
  // 1) GRU recurrence (96+ GFLOP, all WMMA)
  gru_kernel<<<320, 256, 0, stream>>>(obs, oWih, oWhh, bih, bhh, hF, hBf);
  // 2) xp = h @ gat_W           [5120,128] x [128,128]
  gemm_bf16_kernel<<<640, 128, 0, stream>>>(hBf, ogW, nullptr, xpF, nullptr,
                                            5120, 128, 128, 0);
  // 3) GAT attention over 5-node chain -> g (+gat_b)
  gat_attn_kernel<<<80, 256, 0, stream>>>(xpF, attS, attD, gB, gF, gBf);
  // 4) m = relu(g @ W1 + b1)    [5120,128] x [128,128]
  gemm_bf16_kernel<<<640, 128, 0, stream>>>(gBf, oW1, b1, mF, mBf,
                                            5120, 128, 128, 1);
  // 5) t = relu(m @ W2 + b2)    [5120,128] x [128,64]
  gemm_bf16_kernel<<<320, 128, 0, stream>>>(mBf, oW2, b2, tF, nullptr,
                                            5120, 64, 128, 1);
  // 6) actor heads + critic
  heads_kernel<<<84, 256, 0, stream>>>(tF, aW, aB, cW, cB, (float*)d_out);
}
